// MultiHeadLatentAttention_9526237462742
// MI455X (gfx1250) — matrix operand order repaired and checked by reading, not compiled, not run on hardware
//
#include <hip/hip_runtime.h>

// ---------------------------------------------------------------------------
// MLA forward for gfx1250 (MI455X).
//  - all matmuls via v_wmma_f32_16x16x32_f16 (f32 accumulate)
//  - attention K/V tiles staged LDS via TDM tensor_load_to_lds (inline asm),
//    double-buffered, with TDM pad feature providing LDS bank padding
// ---------------------------------------------------------------------------

typedef __attribute__((ext_vector_type(16))) _Float16 v16h;
typedef __attribute__((ext_vector_type(8)))  _Float16 v8h;
typedef __attribute__((ext_vector_type(8)))  float    v8f;
typedef __attribute__((ext_vector_type(4)))  unsigned int u32x4;
typedef __attribute__((ext_vector_type(8)))  unsigned int u32x8;

#define DMODEL 1024
#define NHEADS 16
#define HDIM   64
#define DC     256

__device__ __forceinline__ v16h cat16(v8h lo, v8h hi) {
  return __builtin_shufflevector(lo, hi, 0,1,2,3,4,5,6,7,8,9,10,11,12,13,14,15);
}

// wave32 XOR butterfly via ds_swizzle (group-of-32 mode: {xor[14:10], or[9:5], and[4:0]})
template<int XORMASK>
__device__ __forceinline__ float swz_xor(float v) {
  int i = __builtin_bit_cast(int, v);
  int r = __builtin_amdgcn_ds_swizzle(i, (XORMASK << 10) | 0x1F);
  return __builtin_bit_cast(float, r);
}
__device__ __forceinline__ float redmax16(float v) {
  v = fmaxf(v, swz_xor<1>(v)); v = fmaxf(v, swz_xor<2>(v));
  v = fmaxf(v, swz_xor<4>(v)); v = fmaxf(v, swz_xor<8>(v));
  return v;
}
__device__ __forceinline__ float redsum16(float v) {
  v += swz_xor<1>(v); v += swz_xor<2>(v);
  v += swz_xor<4>(v); v += swz_xor<8>(v);
  return v;
}

// ---------------------------------------------------------------------------
// TDM: 2D tile (tile0 x tile1 elements of 2 bytes) global -> LDS.
// pad_enable with pad_interval=32 DWORDs (128B = one 64-half row) and
// pad_amount=4 DWORDs (16B) => LDS row stride 144B = 72 halves (LDK).
// D# layout per CDNA5 ISA.
// ---------------------------------------------------------------------------
__device__ __forceinline__ void tdm_load_2d_f16(unsigned lds_addr, const void* gptr,
                                                unsigned tensor_d0,
                                                unsigned long long stride0,
                                                unsigned tile0, unsigned tile1)
{
  const unsigned long long ga = (unsigned long long)(uintptr_t)gptr;
  u32x4 g0;
  g0[0] = 1u;                                   // count=1, is_restore=0, gather=0
  g0[1] = lds_addr;                             // LDS byte address
  g0[2] = (unsigned)ga;                         // global_addr[31:0]
  g0[3] = (unsigned)((ga >> 32) & 0x01FFFFFFu)  // global_addr[56:32]
        | (2u << 30);                           // type = 2 ("image")
  u32x8 g1;
  g1[0] = (1u << 16)                            // data_size = 2 bytes
        | (1u << 20)                            // pad_enable
        | (4u << 22)                            // pad_interval: 32 DWORDs
        | (3u << 25);                           // pad_amount: 4 DWORDs
  g1[1] = (tensor_d0 & 0xFFFFu) << 16;          // tensor_dim0[15:0]
  g1[2] = (tensor_d0 >> 16)                     // tensor_dim0[31:16]
        | 0u;                                   // tensor_dim1[15:0] (dim1 = 1<<30)
  g1[3] = (1u << 14)                            // tensor_dim1[31:16] of (1<<30)
        | (tile0 << 16);                        // tile_dim0
  g1[4] = tile1;                                // tile_dim1 (tile_dim2 = 0)
  g1[5] = (unsigned)stride0;                    // tensor_dim0_stride[31:0]
  g1[6] = (unsigned)(stride0 >> 32);            // tensor_dim0_stride[47:32]
  g1[7] = 0u;                                   // tensor_dim1_stride (unused, 2D)
  asm volatile("tensor_load_to_lds %0, %1" :: "s"(g0), "s"(g1) : "memory");
}

__device__ __forceinline__ unsigned lds_addr_of(const void* p) {
  // flat addresses in the LDS aperture map to LDS by truncation to 32 bits
  return (unsigned)(uintptr_t)p;
}

// ---------------------------------------------------------------------------
// Tiled GEMM: C[M,N] = A[M,K](f32) x B[K,N](f32), f16 WMMA, f32 accumulate.
// 256 thr = 8 waves; block tile 128x128, K-step 64; wave tile 32x64.
// MODE 0: f32 row-major out; MODE 1: f16 row-major out;
// MODE 2: f16 out written as V^T laid out [B, H, HDIM, S].
// ---------------------------------------------------------------------------
template<int MODE>
__global__ __launch_bounds__(256)
void gemm_wmma_f16(const float* __restrict__ A, const float* __restrict__ B,
                   void* __restrict__ Cout, int M, int N, int K, int S_dim)
{
  constexpr int LDA = 72;                 // halves; 144 B row stride (16B aligned)
  __shared__ _Float16 As[128 * LDA];      // [row][kk0..63]
  __shared__ _Float16 Bs[128 * LDA];      // [col][kk0..63]  (B staged transposed)

  const int tid   = threadIdx.x;
  const int lane  = tid & 31;
  const int w     = tid >> 5;
  const int wm    = w >> 1;               // 0..3
  const int wn    = w & 1;                // 0..1
  const int l15   = lane & 15;
  const int lhalf = (lane >> 4) & 1;
  const int m0    = blockIdx.y * 128;
  const int n0    = blockIdx.x * 128;

  v8f acc[2][4];
  #pragma unroll
  for (int s = 0; s < 2; ++s)
    #pragma unroll
    for (int nt = 0; nt < 4; ++nt) { v8f z = {}; acc[s][nt] = z; }

  const int lrow = tid & 127;             // row (A) / col (B) for this thread
  const int koff = (tid >> 7) * 32;       // 0 or 32

  for (int k0 = 0; k0 < K; k0 += 64) {
    // ---- stage A tile (f32 -> f16), 32 contiguous K per thread ----
    {
      const float4* ap4 = (const float4*)(A + (size_t)(m0 + lrow) * K + k0 + koff);
      float fv[32];
      #pragma unroll
      for (int v = 0; v < 8; ++v) {
        float4 f = ap4[v];
        fv[4*v+0] = f.x; fv[4*v+1] = f.y; fv[4*v+2] = f.z; fv[4*v+3] = f.w;
      }
      #pragma unroll
      for (int g = 0; g < 4; ++g) {
        v8h hh;
        #pragma unroll
        for (int e = 0; e < 8; ++e) hh[e] = (_Float16)fv[8*g + e];
        *(v8h*)&As[lrow * LDA + koff + 8*g] = hh;
      }
    }
    // ---- stage B tile transposed: Bs[col][kk]; coalesced across lanes ----
    {
      const float* bp = B + (size_t)(k0 + koff) * N + n0 + lrow;
      float fv[32];
      #pragma unroll
      for (int e = 0; e < 32; ++e) fv[e] = bp[(size_t)e * N];
      #pragma unroll
      for (int g = 0; g < 4; ++g) {
        v8h hh;
        #pragma unroll
        for (int e = 0; e < 8; ++e) hh[e] = (_Float16)fv[8*g + e];
        *(v8h*)&Bs[lrow * LDA + koff + 8*g] = hh;
      }
    }
    __syncthreads();

    // ---- compute: 2 K-substeps x (2 M-subtiles x 4 N-subtiles) WMMAs ----
    #pragma unroll
    for (int ks = 0; ks < 2; ++ks) {
      v16h af[2];
      #pragma unroll
      for (int s = 0; s < 2; ++s) {
        const _Float16* p = &As[(wm*32 + s*16 + l15) * LDA + ks*32 + lhalf*8];
        af[s] = cat16(*(const v8h*)p, *(const v8h*)(p + 16));
      }
      #pragma unroll
      for (int nt = 0; nt < 4; ++nt) {
        const _Float16* p = &Bs[(wn*64 + nt*16 + l15) * LDA + ks*32 + lhalf*8];
        v16h bf = cat16(*(const v8h*)p, *(const v8h*)(p + 16));
        #pragma unroll
        for (int s = 0; s < 2; ++s)
          acc[s][nt] = __builtin_amdgcn_wmma_f32_16x16x32_f16(
              false, af[s], false, bf, (short)0, acc[s][nt], false, false);
      }
    }
    __syncthreads();
  }

  // ---- epilogue: C layout lane = (row r + 8*half, col lane&15) ----
  #pragma unroll
  for (int s = 0; s < 2; ++s)
    #pragma unroll
    for (int nt = 0; nt < 4; ++nt)
      #pragma unroll
      for (int r = 0; r < 8; ++r) {
        const int row = m0 + wm*32 + s*16 + r + lhalf*8;
        const int col = n0 + wn*64 + nt*16 + l15;
        const float vv = acc[s][nt][r];
        if constexpr (MODE == 0) {
          ((float*)Cout)[(size_t)row * N + col] = vv;
        } else if constexpr (MODE == 1) {
          ((_Float16*)Cout)[(size_t)row * N + col] = (_Float16)vv;
        } else {
          const int b_   = (row >= S_dim) ? 1 : 0;     // B == 2
          const int spos = row - b_ * S_dim;
          const int h_   = col >> 6;
          const int d_   = col & 63;
          ((_Float16*)Cout)[(((size_t)b_ * NHEADS + h_) * HDIM + d_) * S_dim + spos]
              = (_Float16)vv;
        }
      }
}

// ---------------------------------------------------------------------------
// Flash-style causal attention. Block = 128 thr (4 waves) per (b,h,64-query
// tile). Q in registers. K tile [key][d] and V^T tile [d][key] staged in LDS
// by the Tensor Data Mover, double-buffered (TDM of tile j+1 overlaps WMMA
// of tile j), padded by TDM to a 72-half row stride.
// ---------------------------------------------------------------------------
__global__ __launch_bounds__(128)
void mla_attn_wmma(const _Float16* __restrict__ Qh, const _Float16* __restrict__ Kh,
                   const _Float16* __restrict__ Vt, float* __restrict__ Out, int S)
{
  constexpr int LDK = 72;                   // halves; 144 B stride (TDM-padded)
  __shared__ _Float16 Ks[2][64 * LDK];      // [buf][key][d]
  __shared__ _Float16 Vs[2][64 * LDK];      // [buf][d][key]
  __shared__ _Float16 Ps[4][16 * LDK];      // per-wave P scratch [qrow][key]

  const int tid   = threadIdx.x;
  const int lane  = tid & 31;
  const int w     = tid >> 5;               // wave 0..3
  const int l15   = lane & 15;
  const int lhalf = (lane >> 4) & 1;
  const int qtile = blockIdx.x;
  const int bh    = blockIdx.y;
  const int b     = bh >> 4;
  const int h     = bh & 15;
  const int q0    = qtile * 64;

  // Q A-fragments for this wave's 16 rows (d=64 -> 2 fragments of K=32)
  const size_t qrow = ((size_t)(b * S + q0 + w*16 + l15) * NHEADS + h) * HDIM;
  v16h qf[2];
  #pragma unroll
  for (int kt = 0; kt < 2; ++kt) {
    const _Float16* p = Qh + qrow + kt*32 + lhalf*8;
    qf[kt] = cat16(*(const v8h*)p, *(const v8h*)(p + 16));
  }

  v8f oacc[4];
  #pragma unroll
  for (int dt = 0; dt < 4; ++dt) { v8f z = {}; oacc[dt] = z; }
  float Mr[8], Sr[8];
  #pragma unroll
  for (int r = 0; r < 8; ++r) { Mr[r] = -__builtin_inff(); Sr[r] = 0.0f; }

  // TDM source bases (wave-uniform)
  const _Float16* Kbase = Kh + ((size_t)(b * S) * NHEADS + h) * HDIM; // +j*64*1024
  const _Float16* Vbase = Vt + ((size_t)b * NHEADS + h) * HDIM * (size_t)S; // +j*64

  // prefetch tile j=0 into buffer 0
  if (w == 0) {
    tdm_load_2d_f16(lds_addr_of(&Ks[0][0]), Kbase, HDIM,
                    (unsigned long long)(NHEADS * HDIM), 64u, 64u);
    tdm_load_2d_f16(lds_addr_of(&Vs[0][0]), Vbase, (unsigned)S,
                    (unsigned long long)S, 64u, 64u);
    __builtin_amdgcn_s_wait_tensorcnt(0);
  }
  __syncthreads();

  for (int j = 0; j <= qtile; ++j) {
    const int cur = j & 1;
    // issue TDM for tile j+1 into the other buffer; overlaps with compute
    if (w == 0 && j < qtile) {
      tdm_load_2d_f16(lds_addr_of(&Ks[cur ^ 1][0]),
                      Kbase + (size_t)(j + 1) * 64 * NHEADS * HDIM, HDIM,
                      (unsigned long long)(NHEADS * HDIM), 64u, 64u);
      tdm_load_2d_f16(lds_addr_of(&Vs[cur ^ 1][0]),
                      Vbase + (size_t)(j + 1) * 64, (unsigned)S,
                      (unsigned long long)S, 64u, 64u);
    }
    const _Float16* KsB = &Ks[cur][0];
    const _Float16* VsB = &Vs[cur][0];

    // ---- scores S = Q K^T : 4 N-tiles x 2 K-steps of WMMA ----
    v8f sc[4];
    #pragma unroll
    for (int nt = 0; nt < 4; ++nt) {
      v8f a = {};
      #pragma unroll
      for (int kt = 0; kt < 2; ++kt) {
        const _Float16* p = &KsB[(nt*16 + l15) * LDK + kt*32 + lhalf*8];
        v16h bf = cat16(*(const v8h*)p, *(const v8h*)(p + 16));
        a = __builtin_amdgcn_wmma_f32_16x16x32_f16(
                false, qf[kt], false, bf, (short)0, a, false, false);
      }
      sc[nt] = a;
    }

    // ---- online softmax (per row r; row id = r + 8*lhalf) ----
    const bool diag = (j == qtile);
    #pragma unroll
    for (int r = 0; r < 8; ++r) {
      const int qpos = q0 + w*16 + r + lhalf*8;
      float sv[4];
      float mx = -__builtin_inff();
      #pragma unroll
      for (int nt = 0; nt < 4; ++nt) {
        float s = sc[nt][r] * 0.125f;     // 1/sqrt(64)
        const int kpos = j*64 + nt*16 + l15;
        if (diag && kpos > qpos) s = -__builtin_inff();
        sv[nt] = s;
        mx = fmaxf(mx, s);
      }
      mx = redmax16(mx);
      const float Mnew  = fmaxf(Mr[r], mx);
      const float alpha = __expf(Mr[r] - Mnew);
      float psum = 0.0f;
      #pragma unroll
      for (int nt = 0; nt < 4; ++nt) {
        const float e = __expf(sv[nt] - Mnew);
        psum += e;
        Ps[w][(r + lhalf*8) * LDK + nt*16 + l15] = (_Float16)e;
      }
      psum = redsum16(psum);
      Sr[r] = Sr[r] * alpha + psum;
      Mr[r] = Mnew;
      #pragma unroll
      for (int dt = 0; dt < 4; ++dt) oacc[dt][r] *= alpha;
    }

    // ---- O += P V : reload P as A-fragments (per-wave LDS, in-order DS) ----
    #pragma unroll
    for (int kt = 0; kt < 2; ++kt) {
      const _Float16* ap = &Ps[w][l15 * LDK + kt*32 + lhalf*8];
      v16h af = cat16(*(const v8h*)ap, *(const v8h*)(ap + 16));
      #pragma unroll
      for (int dt = 0; dt < 4; ++dt) {
        const _Float16* bp = &VsB[(dt*16 + l15) * LDK + kt*32 + lhalf*8];
        v16h bf = cat16(*(const v8h*)bp, *(const v8h*)(bp + 16));
        oacc[dt] = __builtin_amdgcn_wmma_f32_16x16x32_f16(
                       false, af, false, bf, (short)0, oacc[dt], false, false);
      }
    }

    // wave 0 drains its TDM ops, then everyone syncs before switching buffers
    if (w == 0) __builtin_amdgcn_s_wait_tensorcnt(0);
    __syncthreads();
  }

  // ---- normalize and write attention output [B,S,H*64] (f32) ----
  #pragma unroll
  for (int r = 0; r < 8; ++r) {
    const float inv = 1.0f / Sr[r];
    const size_t row = (size_t)(b * S + q0 + w*16 + r + lhalf*8) * DMODEL
                     + (size_t)h * HDIM;
    #pragma unroll
    for (int dt = 0; dt < 4; ++dt)
      Out[row + dt*16 + l15] = oacc[dt][r] * inv;
  }
}

// ---------------------------------------------------------------------------
extern "C" void kernel_launch(void* const* d_in, const int* in_sizes, int n_in,
                              void* d_out, int out_size, void* d_ws, size_t ws_size,
                              hipStream_t stream)
{
  (void)n_in; (void)out_size; (void)ws_size;
  const float* x     = (const float*)d_in[0];
  const float* W_dq  = (const float*)d_in[1];
  const float* W_uq  = (const float*)d_in[2];
  const float* W_dkv = (const float*)d_in[3];
  const float* W_uk  = (const float*)d_in[4];
  const float* W_uv  = (const float*)d_in[5];
  const float* W_o   = (const float*)d_in[6];

  const int M = in_sizes[0] / DMODEL;     // B*S = 4096
  const int S = M / 2;                    // B = 2

  // workspace layout (~48 MB)
  float*    c_q  = (float*)d_ws;                          // M*DC f32
  float*    c_kv = c_q + (size_t)M * DC;                  // M*DC f32
  _Float16* Qh   = (_Float16*)(c_kv + (size_t)M * DC);    // M*DMODEL f16 [B,S,H,64]
  _Float16* Kh   = Qh + (size_t)M * DMODEL;               // [B,S,H,64]
  _Float16* Vth  = Kh + (size_t)M * DMODEL;               // V^T: [B,H,64,S]
  float*    attn = (float*)(Vth + (size_t)M * DMODEL);    // M*DMODEL f32

  const dim3 blk(256);
  // latent projections
  gemm_wmma_f16<0><<<dim3(DC/128, M/128), blk, 0, stream>>>(x, W_dq,  c_q,  M, DC, DMODEL, S);
  gemm_wmma_f16<0><<<dim3(DC/128, M/128), blk, 0, stream>>>(x, W_dkv, c_kv, M, DC, DMODEL, S);
  // up-projections straight to f16 for the attention WMMAs (V pre-transposed)
  gemm_wmma_f16<1><<<dim3(DMODEL/128, M/128), blk, 0, stream>>>(c_q,  W_uq, Qh,  M, DMODEL, DC, S);
  gemm_wmma_f16<1><<<dim3(DMODEL/128, M/128), blk, 0, stream>>>(c_kv, W_uk, Kh,  M, DMODEL, DC, S);
  gemm_wmma_f16<2><<<dim3(DMODEL/128, M/128), blk, 0, stream>>>(c_kv, W_uv, Vth, M, DMODEL, DC, S);
  // causal flash attention (TDM-staged, double-buffered K/V tiles)
  mla_attn_wmma<<<dim3(S/64, 2 * NHEADS), dim3(128), 0, stream>>>(Qh, Kh, Vth, attn, S);
  // output projection
  gemm_wmma_f16<0><<<dim3(DMODEL/128, M/128), blk, 0, stream>>>(attn, W_o, (float*)d_out, M, DMODEL, DMODEL, S);
}
